// minGRU_44418551775555
// MI455X (gfx1250) — matrix operand order repaired and checked by reading, not compiled, not run on hardware
//
#include <hip/hip_runtime.h>

// ---------------------------------------------------------------------------
// minGRU (Heinsen log-space scan) for MI455X / gfx1250, wave32 + WMMA.
//   prep : x, w_hg, w_out -> bf16 (weights also transposed to [N][K])
//   GEMM1: hg = x @ w_hg            (v_wmma_f32_16x16x32_bf16, pure-copy staging)
//   scan : h  = exp(a* + cumlogsumexp(lv - a*))   (per-channel sequential)
//   GEMM2: out = h @ w_out          (bf16 WMMA, h produced as bf16 by the scan)
// ---------------------------------------------------------------------------

typedef __attribute__((ext_vector_type(16))) __bf16       v16bf;
typedef __attribute__((ext_vector_type(8)))  float        v8f;
typedef __attribute__((ext_vector_type(4)))  unsigned int u32x4;  // 8 bf16
typedef __attribute__((ext_vector_type(4)))  float        f32x4;

__device__ __forceinline__ __bf16 f2bf(float f) {
  unsigned u = __builtin_bit_cast(unsigned, f);
  unsigned r = u + 0x7FFFu + ((u >> 16) & 1u);   // round-to-nearest-even
  unsigned short h = (unsigned short)(r >> 16);
  return __builtin_bit_cast(__bf16, h);
}

union FragBF {           // 16 bf16 = one WMMA A/B operand (8 VGPRs)
  v16bf v;
  u32x4 q[2];
};

// Block tile 128x128, K-step 64 (two WMMA-K sub-steps). 256 threads = 8 waves
// in a 4x2 grid; each wave owns a 32x64 patch = 2x4 WMMA accumulators.
static constexpr int BM  = 128;
static constexpr int BN  = 128;
static constexpr int BK  = 64;
static constexpr int LDT = 72;   // padded LDS stride (bf16): 36 words, conflict-free

// A: [M][K] bf16 row-major.  BT: [N][K] bf16 (pre-transposed).  C: [M][N] f32.
__global__ __launch_bounds__(256)
void wmma_gemm_bf16(const __bf16* __restrict__ A, const __bf16* __restrict__ BT,
                    float* __restrict__ Cout, int M, int N, int K)
{
  __shared__ __bf16 sA[BM * LDT];   // [row][k]
  __shared__ __bf16 sB[BN * LDT];   // [col][k]

  const int tid  = threadIdx.x;
  const int lane = tid & 31;
  const int wave = tid >> 5;
  const int wm   = (wave >> 1) * 32;   // wave row offset in block tile
  const int wn   = (wave & 1)  * 64;   // wave col offset in block tile
  const int m0   = blockIdx.y * BM;
  const int n0   = blockIdx.x * BN;
  const int l15  = lane & 15;
  const int lhi  = lane >> 4;          // 16-lane half

  v8f acc[2][4];
  for (int i = 0; i < 2; ++i)
    for (int j = 0; j < 4; ++j)
      for (int e = 0; e < 8; ++e) acc[i][j][e] = 0.0f;

  // staging: each thread copies 32 contiguous bf16 (64B) for A and for B
  const int srow = tid >> 1, skb = (tid & 1) * 32;

  for (int k0 = 0; k0 < K; k0 += BK) {
    {
      const u32x4* Ag = (const u32x4*)(A + (size_t)(m0 + srow) * K + (k0 + skb));
      const u32x4* Bg = (const u32x4*)(BT + (size_t)(n0 + srow) * K + (k0 + skb));
      u32x4* da = (u32x4*)&sA[srow * LDT + skb];
      u32x4* db = (u32x4*)&sB[srow * LDT + skb];
      u32x4 a0 = Ag[0], a1 = Ag[1], a2 = Ag[2], a3 = Ag[3];
      u32x4 b0 = Bg[0], b1 = Bg[1], b2 = Bg[2], b3 = Bg[3];
      da[0] = a0; da[1] = a1; da[2] = a2; da[3] = a3;
      db[0] = b0; db[1] = b1; db[2] = b2; db[3] = b3;
    }

    // prefetch next tiles into GL2 while we compute (global_prefetch_b8)
    if (k0 + BK < K) {
      __builtin_prefetch(A + (size_t)(m0 + srow) * K + (k0 + BK + skb), 0, 1);
      __builtin_prefetch(BT + (size_t)(n0 + srow) * K + (k0 + BK + skb), 0, 1);
    }

    __syncthreads();

    for (int ks = 0; ks < BK; ks += 32) {
      // B fragments: elem i <-> K = ks + lhi*16 + i, col fixed
      FragBF bF[4];
      for (int wc = 0; wc < 4; ++wc) {
        const int col = wn + wc * 16 + l15;
        bF[wc].q[0] = *(const u32x4*)&sB[col * LDT + ks + lhi * 16];
        bF[wc].q[1] = *(const u32x4*)&sB[col * LDT + ks + lhi * 16 + 8];
      }
      // A fragments: elem i <-> K = ks + lhi*16 + i, row fixed
      for (int wr = 0; wr < 2; ++wr) {
        const int row = wm + wr * 16 + l15;
        FragBF aF;
        aF.q[0] = *(const u32x4*)&sA[row * LDT + ks + lhi * 16];
        aF.q[1] = *(const u32x4*)&sA[row * LDT + ks + lhi * 16 + 8];
        for (int wc = 0; wc < 4; ++wc) {
          acc[wr][wc] = __builtin_amdgcn_wmma_f32_16x16x32_bf16(
              false, aF.v, false, bF[wc].v, (short)0, acc[wr][wc], false, false);
        }
      }
    }
    __syncthreads();
  }

  // epilogue: C/D layout VGPR r -> M = r + 8*lhi, N = lane&15
  for (int wr = 0; wr < 2; ++wr)
    for (int wc = 0; wc < 4; ++wc) {
      const int col  = n0 + wn + wc * 16 + l15;
      const int rowb = m0 + wm + wr * 16 + lhi * 8;
      float* cp = Cout + (size_t)rowb * N + col;
      for (int r = 0; r < 8; ++r) cp[(size_t)r * N] = acc[wr][wc][r];
    }
}

// ---- prep kernels -----------------------------------------------------------

// elementwise f32 -> bf16, 8 elements per thread (n must be a multiple of 8)
__global__ __launch_bounds__(256)
void convert_f32_to_bf16(const float* __restrict__ in, __bf16* __restrict__ out,
                         size_t n)
{
  const size_t i = ((size_t)blockIdx.x * blockDim.x + threadIdx.x) * 8;
  if (i >= n) return;
  f32x4 a = ((const f32x4*)(in + i))[0];
  f32x4 b = ((const f32x4*)(in + i))[1];
  union { __bf16 h[8]; u32x4 q; } u;
  for (int j = 0; j < 4; ++j) u.h[j]     = f2bf(a[j]);
  for (int j = 0; j < 4; ++j) u.h[4 + j] = f2bf(b[j]);
  *(u32x4*)(out + i) = u.q;
}

// in: [R][C] f32 row-major  ->  out: [C][R] bf16 (transposed), 32x32 LDS tiles
__global__ __launch_bounds__(256)
void transpose_f32_to_bf16(const float* __restrict__ in, __bf16* __restrict__ out,
                           int R, int C)
{
  __shared__ float tile[32][33];
  const int bx = blockIdx.x * 32;   // column base (C dim)
  const int by = blockIdx.y * 32;   // row base    (R dim)
  const int tx = threadIdx.x;       // 0..31
  const int ty = threadIdx.y;       // 0..7
  for (int r = 0; r < 32; r += 8)
    tile[ty + r][tx] = in[(size_t)(by + ty + r) * C + (bx + tx)];
  __syncthreads();
  for (int r = 0; r < 32; r += 8)
    out[(size_t)(bx + ty + r) * R + (by + tx)] = f2bf(tile[tx][ty + r]);
}

// ---- scan -------------------------------------------------------------------

__device__ __forceinline__ float softplusf(float x) {
  return fmaxf(x, 0.0f) + log1pf(__expf(-fabsf(x)));
}

// One thread per (batch, channel): sequential Heinsen scan in log space.
// Running state: a = cumsum(log_coeffs); r = logsumexp_{i<=t}(lv_i - a_i).
__global__ __launch_bounds__(256)
void mingru_scan(const float* __restrict__ hg, __bf16* __restrict__ h,
                 int Bb, int S, int Di)
{
  const int gid = blockIdx.x * blockDim.x + threadIdx.x;
  if (gid >= Bb * Di) return;
  const int b = gid / Di, d = gid % Di;
  const int twoDi = 2 * Di;
  size_t base  = (size_t)b * S * twoDi + d;
  size_t hbase = (size_t)b * S * Di + d;

  float a = 0.0f;
  float r = -__builtin_inff();
  for (int t = 0; t < S; ++t) {
    const float hid  = hg[base];
    const float gate = hg[base + Di];
    const float spg  = softplusf(gate);
    const float lc   = -spg;          // log(1 - z)
    const float lz   = gate - spg;    // -softplus(-gate)
    const float lg   = (hid >= 0.0f) ? __logf(hid + 0.5f) : -softplusf(-hid);
    a += lc;
    const float v  = (lz + lg) - a;
    const float mx = fmaxf(r, v);
    const float mn = fminf(r, v);
    r = mx + log1pf(__expf(mn - mx));           // logaddexp
    h[hbase] = f2bf(__expf(a + r));
    base  += twoDi;
    hbase += Di;
  }
}

// ---- launch -----------------------------------------------------------------

extern "C" void kernel_launch(void* const* d_in, const int* in_sizes, int n_in,
                              void* d_out, int out_size, void* d_ws, size_t ws_size,
                              hipStream_t stream) {
  (void)in_sizes; (void)n_in; (void)out_size; (void)ws_size;
  const float* x     = (const float*)d_in[0];   // [B,S,D]
  const float* w_hg  = (const float*)d_in[1];   // [D, 2*Di]
  const float* w_out = (const float*)d_in[2];   // [Di, D]
  float*       out   = (float*)d_out;           // [B,S,D]

  constexpr int B = 4, S = 4096, D = 2048, Di = 3072;
  constexpr int M  = B * S;         // 16384
  constexpr int N1 = 2 * Di;        // 6144

  // workspace layout (bytes)
  char* ws = (char*)d_ws;
  float*  hg    = (float*)  ws;                                    // M*N1 f32   402.7 MB
  __bf16* hbf   = (__bf16*)(ws + (size_t)M * N1 * 4);              // M*Di bf16  100.7 MB
  __bf16* xbf   = (__bf16*)(ws + (size_t)M * N1 * 4
                               + (size_t)M * Di * 2);              // M*D bf16    67.1 MB
  __bf16* whgT  = (__bf16*)(ws + (size_t)M * N1 * 4
                               + (size_t)M * Di * 2
                               + (size_t)M * D  * 2);              // N1*D bf16   25.2 MB
  __bf16* woutT = (__bf16*)(ws + (size_t)M * N1 * 4
                               + (size_t)M * Di * 2
                               + (size_t)M * D  * 2
                               + (size_t)N1 * D * 2);              // D*Di bf16   12.6 MB

  dim3 blk(256);
  // prep: convert/transpose inputs to bf16 once (conversion off the GEMM path)
  {
    const size_t nx = (size_t)M * D;                               // 33.55M elems
    convert_f32_to_bf16<<<(unsigned)((nx / 8 + 255) / 256), blk, 0, stream>>>(x, xbf, nx);
    transpose_f32_to_bf16<<<dim3(N1 / 32, D / 32), dim3(32, 8), 0, stream>>>(w_hg, whgT, D, N1);
    transpose_f32_to_bf16<<<dim3(D / 32, Di / 32), dim3(32, 8), 0, stream>>>(w_out, woutT, Di, D);
  }
  // GEMM1: hg = x @ w_hg   (M x N1, K = D)
  wmma_gemm_bf16<<<dim3(N1 / BN, M / BM), blk, 0, stream>>>(xbf, whgT, hg, M, N1, D);
  // scan: h (bf16) from hidden/gate halves of hg
  mingru_scan<<<(B * Di + 255) / 256, blk, 0, stream>>>(hg, hbf, B, S, Di);
  // GEMM2: out = h @ w_out (M x D, K = Di)
  wmma_gemm_bf16<<<dim3(D / BN, M / BM), blk, 0, stream>>>(hbf, woutT, out, M, D, Di);
}
